// simpleSNN_58248346468854
// MI455X (gfx1250) — matrix operand order repaired and checked by reading, not compiled, not run on hardware
//
#include <hip/hip_runtime.h>

typedef _Float16 h16;
typedef __attribute__((ext_vector_type(16))) _Float16 v16h;
typedef __attribute__((ext_vector_type(8)))  _Float16 v8h;
typedef __attribute__((ext_vector_type(4)))  _Float16 v4h;
typedef __attribute__((ext_vector_type(8)))  float    v8f;
typedef __attribute__((ext_vector_type(4)))  float    v4f;

#define B_TOT   4096
#define IN_D    3072
#define H1      1536
#define H2      256
#define OUT_D   10
#define T_STEPS 128
#define DT_TAU_MEM 0.1f    // DT * TAU_MEM_INV
#define SYN_DECAY  0.8f    // 1 - DT * TAU_SYN_INV
#define VTH        0.4f

// ---- LDS layout for the time-loop kernel ----
#define V1_STRIDE 1552                 // 1536 + 16 f32 pad (bank-conflict-free rows)
#define Z1_STRIDE 1544                 // 1536 + 8 f16 pad
#define Z2_STRIDE 260
#define Z2_ELEMS  (16 * Z2_STRIDE)                          // one z2 buffer (floats)
#define OFF_V1   0
#define OFF_Z1   (16 * V1_STRIDE * 4)                       // 99328
#define OFF_Z2   (OFF_Z1 + 16 * Z1_STRIDE * 2)              // 148736 (double-buffered)
#define OFF_W3   (OFF_Z2 + 2 * Z2_ELEMS * 4)                // 182016
#define OFF_VL   (OFF_W3 + 10 * 256 * 4)                    // 192256
#define OFF_IL   (OFF_VL + 160 * 4)                         // 192896
#define OFF_OM   (OFF_IL + 160 * 4)                         // 193536
#define SMEM_BYTES (OFF_OM + 160 * 4)                       // 194176

static __device__ inline v16h cat16(v8h lo, v8h hi) {
  return __builtin_shufflevector(lo, hi, 0,1,2,3,4,5,6,7,8,9,10,11,12,13,14,15);
}

// ---------------- elementwise f32 -> f16 ----------------
__global__ void f32_to_f16_kernel(const float* __restrict__ src,
                                  h16* __restrict__ dst, int n) {
  int i = blockIdx.x * 256 + threadIdx.x;
  if (i < n) dst[i] = (h16)src[i];
}

// ---------------- GEMM1: inp1 = xh @ W1h^T + b1 ----------------
__global__ void gemm1_kernel(const h16* __restrict__ A,
                             const h16* __restrict__ Bm,
                             const float* __restrict__ bias,
                             float* __restrict__ C) {
  const int NT = H1 / 16;                       // 96 column tiles
  int wid  = blockIdx.x * 8 + (threadIdx.x >> 5);
  int lane = threadIdx.x & 31;
  int mt = wid / NT, nt = wid % NT;
  int h  = lane >> 4;                           // lane half selects K-subblock
  int nl = lane & 15;                           // row (A) / col (B) within tile
  const h16* arow = A  + (size_t)(mt * 16 + nl) * IN_D;
  const h16* brow = Bm + (size_t)(nt * 16 + nl) * IN_D;
  v8f acc = {};
  for (int ks = 0; ks < IN_D / 32; ++ks) {
    int kb = ks * 32;
    v16h a = cat16(*(const v8h*)(arow + kb + h * 16),
                   *(const v8h*)(arow + kb + h * 16 + 8));
    v16h b = cat16(*(const v8h*)(brow + kb + h * 16),
                   *(const v8h*)(brow + kb + h * 16 + 8));
    acc = __builtin_amdgcn_wmma_f32_16x16x32_f16(false, a, false, b,
                                                 (short)0, acc, false, false);
  }
  int col = nt * 16 + nl;
  float bv = bias[col];
#pragma unroll
  for (int r = 0; r < 8; ++r) {
    int row = mt * 16 + r + 8 * h;              // C layout: VGPR r, lane half h
    C[(size_t)row * H1 + col] = acc[r] + bv;
  }
}

// ---- phase 3 helper: a3 = z2 @ W3^T + b3, LI cell, running max ----
static __device__ inline void li_readout(const float* __restrict__ z2buf,
                                         const float* __restrict__ w3s,
                                         float* __restrict__ vls,
                                         float* __restrict__ ils,
                                         float* __restrict__ outm,
                                         int tid, int m_rec, int o_rec,
                                         float b3v, const float* __restrict__ nzp,
                                         int t) {
  float s = b3v;
  const v4f* zr = (const v4f*)(z2buf + m_rec * Z2_STRIDE);
  const v4f* wr = (const v4f*)(w3s + o_rec * 256);
#pragma unroll 8
  for (int n4 = 0; n4 < 64; ++n4) {
    v4f zv = zr[n4], wv = wr[n4];
    s += zv[0]*wv[0] + zv[1]*wv[1] + zv[2]*wv[2] + zv[3]*wv[3];
  }
  float vl = vls[tid], il = ils[tid];
  vl = vl + DT_TAU_MEM * (il - vl);       // integrates OLD il
  il = il * SYN_DECAY + s;
  vls[tid] = vl; ils[tid] = il;
  float rec = vl + 0.001f * nzp[(size_t)t * B_TOT * OUT_D];
  outm[tid] = fmaxf(outm[tid], rec);
}

// ---------------- persistent time-loop kernel ----------------
// One workgroup (256 thr / 8 waves) owns 16 batch rows for all 128 steps.
// v1 in LDS; i1 closed-form; v2/i2 in accumulator lanes; z2 double-buffered
// so the LI readout of step t-1 overlaps the LIF-1 update of step t.
__global__ void snn_time_loop_kernel(const float* __restrict__ inp1,   // [4096][1536]
                                     const h16*  __restrict__ W2h,     // [256][1536]
                                     const float* __restrict__ b2,     // [256]
                                     const float* __restrict__ W3,     // [10][256]
                                     const float* __restrict__ b3,     // [10]
                                     const float* __restrict__ noise,  // [128][4096][10]
                                     float* __restrict__ out)          // [4096][10]
{
  extern __shared__ char smem[];
  float* v1s  = (float*)(smem + OFF_V1);   // [16][V1_STRIDE]
  h16*   z1s  = (h16*)  (smem + OFF_Z1);   // [16][Z1_STRIDE]
  float* z2s  = (float*)(smem + OFF_Z2);   // 2 x [16][Z2_STRIDE]
  float* w3s  = (float*)(smem + OFF_W3);   // [10][256]
  float* vls  = (float*)(smem + OFF_VL);   // [160]
  float* ils  = (float*)(smem + OFF_IL);   // [160]
  float* outm = (float*)(smem + OFF_OM);   // [160]

  const int tid  = threadIdx.x;
  const int row0 = blockIdx.x * 16;

  // ---- init ----
  for (int i = tid; i < 16 * V1_STRIDE; i += 256) v1s[i] = 0.0f;
  for (int i = tid; i < 10 * 256;       i += 256) w3s[i] = W3[i];
  if (tid < 160) { vls[tid] = 0.0f; ils[tid] = 0.0f; outm[tid] = -__builtin_inff(); }

  // ---- wave/lane mapping for GEMM2: wave w owns n-tiles {w, w+8} ----
  const int wv   = tid >> 5;
  const int lane = tid & 31;
  const int h    = lane >> 4;
  const int nl   = lane & 15;
  const int n0   = wv * 16 + nl;
  const int n1   = n0 + 128;
  const float b2_0 = b2[n0];
  const float b2_1 = b2[n1];
  const h16* b0row = W2h + (size_t)n0 * H1;
  const h16* b1row = W2h + (size_t)n1 * H1;
  const h16* arow  = z1s + nl * Z1_STRIDE;

  v8f acc0 = {}, acc1 = {};
  float v2r[2][8] = {{0}}, i2r[2][8] = {{0}};

  // ---- phase-1 mapping (float4): thread m = tid/16, k = 4*(tid%16) + 64*j ----
  const int p1_m  = tid >> 4;
  const int p1_c0 = (tid & 15) * 4;
  const float* inp_row = inp1 + (size_t)(row0 + p1_m) * H1;
  float* v1row = v1s + p1_m * V1_STRIDE;
  h16*   z1row = z1s + p1_m * Z1_STRIDE;

  // ---- phase-3 mapping ----
  const int m_rec = tid / 10, o_rec = tid % 10;
  const float b3v = (tid < 160) ? b3[o_rec] : 0.0f;
  const float* nzp = (tid < 160)
      ? noise + (size_t)(row0 + m_rec) * OUT_D + o_rec : noise;

  float powv = 1.0f;   // 0.8^t
  int zb = 0;          // z2 buffer phase-2 writes this step
  __syncthreads();

#pragma unroll 1
  for (int t = 0; t < T_STEPS; ++t) {
    // ---- phase 1: LIF layer 1 (i1 closed-form), vectorized ----
    const float syn_s = (1.0f - powv) * 5.0f;     // (1-0.8^t)/0.2
#pragma unroll 4
    for (int j = 0; j < H1 / 64; ++j) {
      int k = p1_c0 + j * 64;
      v4f cur4 = *(const v4f*)(inp_row + k);
      v4f vv   = *(const v4f*)(v1row + k);
      v4h z4;
#pragma unroll
      for (int c = 0; c < 4; ++c) {
        float cur = cur4[c] * syn_s;
        float vd  = vv[c] + DT_TAU_MEM * (cur - vv[c]);
        bool  sp  = vd > VTH;
        vv[c] = sp ? 0.0f : vd;
        z4[c] = (h16)(sp ? 1.0f : 0.0f);
      }
      *(v4f*)(v1row + k) = vv;
      *(v4h*)(z1row + k) = z4;
    }

    // ---- overlapped phase 3 for step t-1 (reads the other z2 buffer) ----
    if (t > 0 && tid < 160)
      li_readout(z2s + (zb ^ 1) * Z2_ELEMS, w3s, vls, ils, outm,
                 tid, m_rec, o_rec, b3v, nzp, t - 1);
    __syncthreads();

    // ---- phase 2: a2 = z1 @ W2^T via WMMA f16 (z1 exact in f16) ----
#pragma unroll 4
    for (int ks = 0; ks < H1 / 32; ++ks) {
      int kb = ks * 32;
      v16h a  = cat16(*(const v8h*)(arow + kb + h * 16),
                      *(const v8h*)(arow + kb + h * 16 + 8));
      v16h bb0 = cat16(*(const v8h*)(b0row + kb + h * 16),
                       *(const v8h*)(b0row + kb + h * 16 + 8));
      v16h bb1 = cat16(*(const v8h*)(b1row + kb + h * 16),
                       *(const v8h*)(b1row + kb + h * 16 + 8));
      __builtin_prefetch((const void*)(b0row + kb + 128), 0, 1);
      __builtin_prefetch((const void*)(b1row + kb + 128), 0, 1);
      acc0 = __builtin_amdgcn_wmma_f32_16x16x32_f16(false, a, false, bb0,
                                                    (short)0, acc0, false, false);
      acc1 = __builtin_amdgcn_wmma_f32_16x16x32_f16(false, a, false, bb1,
                                                    (short)0, acc1, false, false);
    }

    // ---- layer-2 LIF in accumulator registers ----
    float* z2w = z2s + zb * Z2_ELEMS;
#pragma unroll
    for (int r = 0; r < 8; ++r) {
      int m = r + 8 * h;
      {
        float a2 = acc0[r] + b2_0;
        float vd = v2r[0][r] + DT_TAU_MEM * (i2r[0][r] - v2r[0][r]);
        i2r[0][r] = i2r[0][r] * SYN_DECAY + a2;
        v2r[0][r] = (vd > VTH) ? 0.0f : vd;
        z2w[m * Z2_STRIDE + n0] = (vd > VTH) ? 1.0f : 0.0f;
        acc0[r] = 0.0f;
      }
      {
        float a2 = acc1[r] + b2_1;
        float vd = v2r[1][r] + DT_TAU_MEM * (i2r[1][r] - v2r[1][r]);
        i2r[1][r] = i2r[1][r] * SYN_DECAY + a2;
        v2r[1][r] = (vd > VTH) ? 0.0f : vd;
        z2w[m * Z2_STRIDE + n1] = (vd > VTH) ? 1.0f : 0.0f;
        acc1[r] = 0.0f;
      }
    }
    __syncthreads();

    zb ^= 1;
    powv *= SYN_DECAY;
  }

  // ---- drain: phase 3 for the final step ----
  if (tid < 160) {
    li_readout(z2s + (zb ^ 1) * Z2_ELEMS, w3s, vls, ils, outm,
               tid, m_rec, o_rec, b3v, nzp, T_STEPS - 1);
    out[(size_t)(row0 + m_rec) * OUT_D + o_rec] = outm[tid];
  }
}

// ---------------- host ----------------
extern "C" void kernel_launch(void* const* d_in, const int* in_sizes, int n_in,
                              void* d_out, int out_size, void* d_ws, size_t ws_size,
                              hipStream_t stream) {
  (void)in_sizes; (void)n_in; (void)out_size; (void)ws_size;
  const float* x     = (const float*)d_in[0];
  const float* W1    = (const float*)d_in[1];
  const float* b1    = (const float*)d_in[2];
  const float* W2    = (const float*)d_in[3];
  const float* b2    = (const float*)d_in[4];
  const float* W3    = (const float*)d_in[5];
  const float* b3    = (const float*)d_in[6];
  const float* noise = (const float*)d_in[7];
  float* out = (float*)d_out;

  char* ws = (char*)d_ws;
  h16*   xh   = (h16*)ws;                                         // 4096*3072 f16
  h16*   w1h  = (h16*)(ws + (size_t)B_TOT * IN_D * 2);            // 1536*3072 f16
  h16*   w2h  = (h16*)(ws + (size_t)B_TOT * IN_D * 2
                          + (size_t)H1 * IN_D * 2);               // 256*1536 f16
  float* inp1 = (float*)(ws + (size_t)B_TOT * IN_D * 2
                            + (size_t)H1 * IN_D * 2
                            + (size_t)H2 * H1 * 2);               // 4096*1536 f32

  f32_to_f16_kernel<<<(B_TOT * IN_D + 255) / 256, 256, 0, stream>>>(x,  xh,  B_TOT * IN_D);
  f32_to_f16_kernel<<<(H1 * IN_D   + 255) / 256, 256, 0, stream>>>(W1, w1h, H1 * IN_D);
  f32_to_f16_kernel<<<(H2 * H1     + 255) / 256, 256, 0, stream>>>(W2, w2h, H2 * H1);

  // 256 M-tiles x 96 N-tiles, 8 waves per block
  gemm1_kernel<<<(B_TOT / 16) * (H1 / 16) / 8, 256, 0, stream>>>(xh, w1h, b1, inp1);

  hipFuncSetAttribute((const void*)snn_time_loop_kernel,
                      hipFuncAttributeMaxDynamicSharedMemorySize, SMEM_BYTES);
  snn_time_loop_kernel<<<B_TOT / 16, 256, SMEM_BYTES, stream>>>(
      inp1, w2h, b2, W3, b3, noise, out);
}
